// DadaDyGNN_19370302505376
// MI455X (gfx1250) — matrix operand order repaired and checked by reading, not compiled, not run on hardware
//
#include <hip/hip_runtime.h>

// ---------------------------------------------------------------------------
// DyGNN forward for MI455X (gfx1250, wave32, WMMA bf16 16x16x32).
// ---------------------------------------------------------------------------

typedef __attribute__((ext_vector_type(16))) __bf16 v16bf;
typedef __attribute__((ext_vector_type(8)))  float  v8f;
typedef __attribute__((ext_vector_type(4)))  unsigned int u32x4;
typedef __attribute__((ext_vector_type(4)))  float  f32x4;

union Frag {
  v16bf v;
  u32x4 q[2];
};
static_assert(sizeof(Frag) == 32, "frag size");

union Pack16 {            // 16 bf16 values = 32 bytes
  unsigned short u[16];
  u32x4 q[2];
};

__device__ __forceinline__ unsigned short f2bf(float f) {
  unsigned u = __float_as_uint(f);
  unsigned r = u + 0x7FFFu + ((u >> 16) & 1u);   // round to nearest even
  return (unsigned short)(r >> 16);
}

__device__ __forceinline__ float wred(float s) {
#pragma unroll
  for (int o = 16; o > 0; o >>= 1) s += __shfl_xor(s, o, 32);
  return s;
}

// ---------------------------------------------------------------------------
// Weight prep: f32 [K][N] -> bf16 [Kpad][N] (zero padded rows).
// ---------------------------------------------------------------------------
__global__ void conv_bf16_pad(const float* __restrict__ W, unsigned short* __restrict__ D,
                              int K, int N, int Kpad) {
  long long i = (long long)blockIdx.x * blockDim.x + threadIdx.x;
  long long tot = (long long)Kpad * N;
  if (i >= tot) return;
  int k = (int)(i / N);
  int n = (int)(i % N);
  float v = (k < K) ? W[(long long)k * N + n] : 0.0f;
  D[i] = f2bf(v);
}

// v_j = W_g @ a[j*256 : (j+1)*256],  j = 0..2  (for attention-score factoring)
__global__ void wg_a_kernel(const float* __restrict__ Wg, const float* __restrict__ a,
                            float* __restrict__ v012) {
  int j = blockIdx.x;   // 0..2
  int e = threadIdx.x;  // 0..255
  const float* row = Wg + (long long)e * 256;
  const float* aj = a + j * 256;
  float s = 0.f;
  for (int m = 0; m < 256; ++m) s += row[m] * aj[m];
  v012[j * 256 + e] = s;
}

// ---------------------------------------------------------------------------
// Message attention: per (b), scores over K=64 neighbors via precomputed
// v0/v1/v2, softmax, then weighted neighbor sums u_ins/u_del -> U rows.
// ---------------------------------------------------------------------------
__global__ __launch_bounds__(256) void message_kernel(
    const float* __restrict__ emb, const float* __restrict__ v012,
    const float* __restrict__ timestamps, const int* __restrict__ idxs,
    const int* __restrict__ nbrs, const float* __restrict__ ins_t,
    const float* __restrict__ del_t, float* __restrict__ U, int uoff) {
  int b = blockIdx.x;
  int tid = threadIdx.x, lane = tid & 31, wave = tid >> 5;
  __shared__ int nbs[64];
  __shared__ float d1s[64], d2s[64], pis[64], pds[64], wis[64], wds[64];
  __shared__ float c0sh;
  if (tid < 64) nbs[tid] = nbrs[b * 64 + tid];
  __syncthreads();
  float ts = timestamps[b];
  if (wave == 0) {                    // c0 = emb[idx] . (W_g@a0)
    const float* er = emb + (long long)idxs[b] * 256;
    float s = 0.f;
    for (int e = lane; e < 256; e += 32) s += er[e] * v012[e];
    s = wred(s);
    if (lane == 0) c0sh = s;
  }
  for (int kk = 0; kk < 8; ++kk) {    // each wave: 8 neighbors
    int k = wave * 8 + kk;
    const float* nr = emb + (long long)nbs[k] * 256;
    float s1 = 0.f, s2 = 0.f;
    for (int e = lane; e < 256; e += 32) {
      float x = nr[e];
      s1 += x * v012[256 + e];
      s2 += x * v012[512 + e];
    }
    s1 = wred(s1); s2 = wred(s2);
    if (lane == 0) {
      d1s[k] = s1; d2s[k] = s2;
      pis[k] = 1.f / (1.f + (ts - ins_t[b * 64 + k]));
      pds[k] = 1.f / (1.f + (ts - del_t[b * 64 + k]));
    }
  }
  __syncthreads();
  if (tid == 0) {                     // leaky-relu + softmax over 64
    float c0 = c0sh, mx = -1e30f;
    for (int k = 0; k < 64; ++k) {
      float v = c0 + pis[k] * d1s[k] - pds[k] * d2s[k];
      v = (v > 0.f) ? v : 0.2f * v;
      d1s[k] = v;
      mx = fmaxf(mx, v);
    }
    float sum = 0.f;
    for (int k = 0; k < 64; ++k) { float e = expf(d1s[k] - mx); d1s[k] = e; sum += e; }
    float inv = 1.f / sum;
    for (int k = 0; k < 64; ++k) {
      float a = d1s[k] * inv;
      wis[k] = a * pis[k];
      wds[k] = -a * pds[k];
    }
  }
  __syncthreads();
  int e = tid;                        // weighted neighbor sums (coalesced)
  float ui = 0.f, ud = 0.f;
  for (int k = 0; k < 64; ++k) {
    float x = emb[(long long)nbs[k] * 256 + e];
    ui += wis[k] * x;
    ud += wds[k] * x;
  }
  long long r0 = ((long long)4 * b + uoff) * 256;
  U[r0 + e] = ui;
  U[r0 + 256 + e] = ud;
}

// ---------------------------------------------------------------------------
// Generic bf16 WMMA GEMM: C = act( A @ Wbf ).
// A assembled on the fly from up to two f32 row sources (optionally
// index-gathered) + broadcast tail scalar column.
// Block = 256 threads (8 waves); tile 128x64; K step 32.
// Wave w owns rows [w*16, w*16+16) x all 64 cols (4 WMMAs sharing A frag).
// ---------------------------------------------------------------------------
struct GemmArgs {
  const float* a0; const int* i0; int c0; int ld0;
  const float* a1; const int* i1; int c1; int ld1;
  const float* tail; int taildiv;       // appended scalar column at k == c0+c1
  const unsigned short* bt;             // bf16 weights, [Kpad][N] row-major
  float* c;
  int M, N, Kpad;
  int outGroup; long long outGroupStride;  // off = (r/g)*gs + (r%g)*N
  int act;                              // 0 none, 1 relu, 2 tanh
};

__global__ __launch_bounds__(256) void gemm_bf16_wmma(GemmArgs g) {
  __shared__ __align__(16) unsigned short As[128][40];  // 128x32 + pad
  __shared__ __align__(16) unsigned short Bs[32][64];   // 32x64
  int mBase = blockIdx.x * 128;
  int nBase = blockIdx.y * 64;
  int tid = threadIdx.x, lane = tid & 31, wave = tid >> 5;
  int l16 = lane & 15;
  int kb = (lane >> 4) * 8;             // A-frag K-base per lane half

  // A staging assignment: thread -> (row, 16-col half)
  int lr = tid >> 1;                    // 0..127
  int lc = (tid & 1) * 16;              // 0 or 16
  int grow = mBase + lr;
  // B staging assignment: thread -> (row, 8-col chunk)
  int br = tid >> 3;                    // 0..31
  int bc = (tid & 7) * 8;               // 0..56

  int ar0 = 0, ar1 = 0;
  float tv = 0.f;
  bool inM = (grow < g.M);
  if (inM) {
    ar0 = g.i0 ? g.i0[grow] : grow;
    if (g.a1) ar1 = g.i1 ? g.i1[grow] : grow;
    if (g.tail) tv = g.tail[grow / g.taildiv];
  }

  v8f acc[4] = {};

  for (int kk = 0; kk < g.Kpad; kk += 32) {
    // ---- stage A tile (128x32) ----
    int k0 = kk + lc;                   // this thread covers [k0, k0+16)
    Pack16 pk;
    if (!inM) {
#pragma unroll
      for (int j = 0; j < 16; ++j) pk.u[j] = 0;
    } else if (k0 + 16 <= g.c0) {
      const float* s = g.a0 + (long long)ar0 * g.ld0 + k0;
      const f32x4* sv = reinterpret_cast<const f32x4*>(s);
#pragma unroll
      for (int q = 0; q < 4; ++q) {
        f32x4 f = sv[q];
#pragma unroll
        for (int j = 0; j < 4; ++j) pk.u[q * 4 + j] = f2bf(f[j]);
      }
    } else if (k0 >= g.c0 && k0 + 16 <= g.c0 + g.c1) {
      const float* s = g.a1 + (long long)ar1 * g.ld1 + (k0 - g.c0);
      const f32x4* sv = reinterpret_cast<const f32x4*>(s);
#pragma unroll
      for (int q = 0; q < 4; ++q) {
        f32x4 f = sv[q];
#pragma unroll
        for (int j = 0; j < 4; ++j) pk.u[q * 4 + j] = f2bf(f[j]);
      }
    } else {
      // rare straddling tile (concat boundary, tail flag, zero pad)
#pragma unroll
      for (int j = 0; j < 16; ++j) {
        int k = k0 + j;
        float v = 0.f;
        if (k < g.c0)                       v = g.a0[(long long)ar0 * g.ld0 + k];
        else if (k < g.c0 + g.c1)           v = g.a1[(long long)ar1 * g.ld1 + (k - g.c0)];
        else if (g.tail && k == g.c0 + g.c1) v = tv;
        pk.u[j] = f2bf(v);
      }
    }
    {
      u32x4* dst = reinterpret_cast<u32x4*>(&As[lr][lc]);
      dst[0] = pk.q[0];
      dst[1] = pk.q[1];
    }
    // ---- stage B tile (32x64), coalesced ----
    {
      const u32x4* src =
          reinterpret_cast<const u32x4*>(g.bt + (long long)(kk + br) * g.N + nBase + bc);
      *reinterpret_cast<u32x4*>(&Bs[br][bc]) = *src;
    }
    __syncthreads();

    // ---- compute: 1 A frag + 4 B frags -> 4 WMMAs ----
    Frag af;
    const u32x4* aq = reinterpret_cast<const u32x4*>(&As[wave * 16 + l16][kb]);
    af.q[0] = aq[0];
    af.q[1] = aq[2];
#pragma unroll
    for (int t = 0; t < 4; ++t) {
      Frag bf;
      Pack16 pb;
#pragma unroll
      for (int s = 0; s < 16; ++s) {
        int k = (s < 8) ? (kb + s) : (16 + kb + (s - 8));
        pb.u[s] = Bs[k][t * 16 + l16];
      }
      bf.q[0] = pb.q[0];
      bf.q[1] = pb.q[1];
      acc[t] = __builtin_amdgcn_wmma_f32_16x16x32_bf16(false, af.v, false, bf.v,
                                                       (short)0, acc[t], false, false);
    }
    __syncthreads();
  }

  // ---- epilogue: activation + grouped-row remap store ----
  int row0 = mBase + wave * 16 + (lane >> 4) * 8;
#pragma unroll
  for (int v = 0; v < 8; ++v) {
    int row = row0 + v;
    if (row < g.M) {
      long long off = (long long)(row / g.outGroup) * g.outGroupStride +
                      (long long)(row % g.outGroup) * g.N;
#pragma unroll
      for (int t = 0; t < 4; ++t) {
        float x = acc[t][v];
        if (g.act == 1)      x = fmaxf(x, 0.f);
        else if (g.act == 2) x = tanhf(x);
        g.c[off + nBase + t * 16 + l16] = x;
      }
    }
  }
}

// ---------------------------------------------------------------------------
// Scatter: mem[idxs[b]] = rows[b]
// ---------------------------------------------------------------------------
__global__ void scatter_rows(float* __restrict__ mem, const int* __restrict__ idxs,
                             const float* __restrict__ rows) {
  int b = blockIdx.x;
  mem[(long long)idxs[b] * 256 + threadIdx.x] = rows[(long long)b * 256 + threadIdx.x];
}

// ---------------------------------------------------------------------------
// Update attention: score_k = (pi-pd)*dot(h1_b, mem[nbr_k]) + EPS, softmax,
// changed[b,k,:] = att_k * h1_b
// ---------------------------------------------------------------------------
__global__ __launch_bounds__(256) void upd_att_kernel(
    const float* __restrict__ mem, const float* __restrict__ h1,
    const float* __restrict__ timestamps, const int* __restrict__ nbrs,
    const float* __restrict__ ins_t, const float* __restrict__ del_t,
    float* __restrict__ changed) {
  int b = blockIdx.x;
  int tid = threadIdx.x, lane = tid & 31, wave = tid >> 5;
  __shared__ float h1s[256];
  __shared__ float sc[32];
  __shared__ int nbs[32];
  h1s[tid] = h1[(long long)b * 256 + tid];
  if (tid < 32) nbs[tid] = nbrs[b * 32 + tid];
  __syncthreads();
  float ts = timestamps[b];
  for (int kk = 0; kk < 4; ++kk) {
    int k = wave * 4 + kk;
    const float* nr = mem + (long long)nbs[k] * 256;
    float s = 0.f;
    for (int e = lane; e < 256; e += 32) s += h1s[e] * nr[e];
    s = wred(s);
    if (lane == 0) {
      float pi = 1.f / (1.f + (ts - ins_t[b * 32 + k]));
      float pd = 1.f / (1.f + (ts - del_t[b * 32 + k]));
      sc[k] = (pi - pd) * s + 1e-10f;
    }
  }
  __syncthreads();
  if (tid == 0) {
    float mx = -1e30f;
    for (int k = 0; k < 32; ++k) mx = fmaxf(mx, sc[k]);
    float sum = 0.f;
    for (int k = 0; k < 32; ++k) { float e = expf(sc[k] - mx); sc[k] = e; sum += e; }
    float inv = 1.f / sum;
    for (int k = 0; k < 32; ++k) sc[k] *= inv;
  }
  __syncthreads();
  float h = h1s[tid];
  for (int k = 0; k < 32; ++k)
    changed[((long long)b * 32 + k) * 256 + tid] = sc[k] * h;
}

// ---------------------------------------------------------------------------
// prob = sigmoid(dot(y_row, W2)); if prob >= 0.5: mem[flat[r]] = changed2[r]
// ---------------------------------------------------------------------------
__global__ __launch_bounds__(256) void upd_apply_kernel(
    float* __restrict__ mem, const float* __restrict__ y,
    const float* __restrict__ w2, const float* __restrict__ changed2,
    const int* __restrict__ flat) {
  long long r = blockIdx.x;
  int tid = threadIdx.x, lane = tid & 31, wave = tid >> 5;
  __shared__ float red[8];
  float s = y[r * 256 + tid] * w2[tid];
  s = wred(s);
  if (lane == 0) red[wave] = s;
  __syncthreads();
  if (tid == 0) {
    float t = 0.f;
    for (int i = 0; i < 8; ++i) t += red[i];
    red[0] = t;
  }
  __syncthreads();
  float prob = 1.f / (1.f + expf(-red[0]));
  if (prob >= 0.5f)
    mem[(long long)flat[r] * 256 + tid] = changed2[r * 256 + tid];
}

// ---------------------------------------------------------------------------
// Final: pos/neg link probabilities.
// ---------------------------------------------------------------------------
__global__ __launch_bounds__(256) void final_kernel(
    const float* __restrict__ mem, const int* __restrict__ si, const int* __restrict__ di,
    const int* __restrict__ nsi, const int* __restrict__ ndi,
    float* __restrict__ out, int B) {
  int b = blockIdx.x;
  int tid = threadIdx.x, lane = tid & 31, wave = tid >> 5;
  __shared__ float red[16];
  const float* ps = mem + (long long)si[b] * 256;
  const float* pd = mem + (long long)di[b] * 256;
  const float* ns = mem + (long long)nsi[b] * 256;
  const float* nd = mem + (long long)ndi[b] * 256;
  float s1 = ps[tid] * pd[tid];
  float s2 = ns[tid] * nd[tid];
  s1 = wred(s1); s2 = wred(s2);
  if (lane == 0) { red[wave] = s1; red[8 + wave] = s2; }
  __syncthreads();
  if (tid == 0) {
    float t1 = 0.f, t2 = 0.f;
    for (int i = 0; i < 8; ++i) { t1 += red[i]; t2 += red[8 + i]; }
    out[b] = 1.f / (1.f + expf(-t1));
    out[B + b] = 1.f / (1.f + expf(-t2));
    if (b == 0) { out[2 * B] = 0.f; out[2 * B + 1] = 0.f; }
  }
}

// ---------------------------------------------------------------------------
extern "C" void kernel_launch(void* const* d_in, const int* in_sizes, int n_in,
                              void* d_out, int out_size, void* d_ws, size_t ws_size,
                              hipStream_t stream) {
  (void)in_sizes; (void)n_in; (void)out_size; (void)ws_size;
  const int B = 2000, KU = 32, E = 256, NN = 100001;

  const float* emb       = (const float*)d_in[0];
  const float* edge_feat = (const float*)d_in[1];
  const float* W_g       = (const float*)d_in[2];
  const float* a_vec     = (const float*)d_in[3];
  const float* W_e       = (const float*)d_in[4];
  const float* W_uc      = (const float*)d_in[5];
  const float* W_un      = (const float*)d_in[6];
  const float* W_p       = (const float*)d_in[7];
  const float* W_1       = (const float*)d_in[8];
  const float* W_2       = (const float*)d_in[9];
  const float* timestamps = (const float*)d_in[10];
  const float* flags     = (const float*)d_in[11];
  const float* ins_src   = (const float*)d_in[12];
  const float* del_src   = (const float*)d_in[13];
  const float* ins_dst   = (const float*)d_in[14];
  const float* del_dst   = (const float*)d_in[15];
  const float* u_ins_src = (const float*)d_in[16];
  const float* u_del_src = (const float*)d_in[17];
  const float* u_ins_dst = (const float*)d_in[18];
  const float* u_del_dst = (const float*)d_in[19];
  const int* src_idxs     = (const int*)d_in[20];
  const int* dst_idxs     = (const int*)d_in[21];
  const int* neg_src_idxs = (const int*)d_in[22];
  const int* neg_dst_idxs = (const int*)d_in[23];
  const int* edge_idxs    = (const int*)d_in[24];
  const int* nbrs_src     = (const int*)d_in[25];
  const int* nbrs_dst     = (const int*)d_in[26];
  const int* unbrs_src    = (const int*)d_in[27];
  const int* unbrs_dst    = (const int*)d_in[28];

  // ---- workspace carve ----
  char* p = (char*)d_ws;
  auto alloc = [&](size_t bytes) -> char* {
    char* r = p;
    p += (bytes + 255) & ~(size_t)255;
    return r;
  };
  float*          mem   = (float*)alloc((size_t)NN * E * 4);
  unsigned short* Wg_b  = (unsigned short*)alloc((size_t)256 * 256 * 2);
  unsigned short* We_b  = (unsigned short*)alloc((size_t)192 * 1024 * 2);
  unsigned short* Wuc_b = (unsigned short*)alloc((size_t)2304 * 256 * 2);
  unsigned short* Wp_b  = (unsigned short*)alloc((size_t)2048 * 256 * 2);
  unsigned short* W1_b  = (unsigned short*)alloc((size_t)544 * 256 * 2);
  unsigned short* Wun_b = (unsigned short*)alloc((size_t)544 * 256 * 2);
  float* v012    = (float*)alloc((size_t)3 * 256 * 4);
  float* U       = (float*)alloc((size_t)4 * B * 256 * 4);
  float* h_full  = (float*)alloc((size_t)B * 2048 * 4);
  float* up_src  = (float*)alloc((size_t)B * 256 * 4);
  float* up_dst  = (float*)alloc((size_t)B * 256 * 4);
  float* h1      = (float*)alloc((size_t)B * 256 * 4);
  float* changed = (float*)alloc((size_t)B * KU * 256 * 4);
  float* ybuf    = (float*)alloc((size_t)B * KU * 256 * 4);
  float* chg2    = (float*)alloc((size_t)B * KU * 256 * 4);

  // ---- mem = copy of emb (do not mutate inputs) ----
  hipMemcpyAsync(mem, emb, (size_t)NN * E * 4, hipMemcpyDeviceToDevice, stream);

  // ---- weight prep (bf16, K zero-padded to multiples of 32) ----
  auto conv = [&](const float* W, unsigned short* D, int K, int N, int Kpad) {
    long long tot = (long long)Kpad * N;
    conv_bf16_pad<<<dim3((unsigned)((tot + 255) / 256)), dim3(256), 0, stream>>>(W, D, K, N, Kpad);
  };
  conv(W_g,  Wg_b,  256,  256,  256);
  conv(W_e,  We_b,  172, 1024,  192);
  conv(W_uc, Wuc_b, 2304, 256, 2304);
  conv(W_p,  Wp_b,  2048, 256, 2048);
  conv(W_1,  W1_b,   513, 256,  544);
  conv(W_un, Wun_b,  513, 256,  544);
  wg_a_kernel<<<dim3(3), dim3(256), 0, stream>>>(W_g, a_vec, v012);

  // ---- message attention (reads original emb) ----
  message_kernel<<<dim3(B), dim3(256), 0, stream>>>(
      emb, v012, timestamps, src_idxs, nbrs_src, ins_src, del_src, U, 0);
  message_kernel<<<dim3(B), dim3(256), 0, stream>>>(
      emb, v012, timestamps, dst_idxs, nbrs_dst, ins_dst, del_dst, U, 2);

  auto launch_gemm = [&](const GemmArgs& g) {
    dim3 grid((unsigned)((g.M + 127) / 128), (unsigned)(g.N / 64));
    gemm_bf16_wmma<<<grid, dim3(256), 0, stream>>>(g);
  };

  // GEMM1: h_full[:, :1024] = relu(U @ W_g), rows (b,4) -> grouped remap
  {
    GemmArgs g{};
    g.a0 = U; g.i0 = nullptr; g.c0 = 256; g.ld0 = 256;
    g.a1 = nullptr; g.i1 = nullptr; g.c1 = 0; g.ld1 = 0;
    g.tail = nullptr; g.taildiv = 1;
    g.bt = Wg_b; g.c = h_full;
    g.M = 4 * B; g.N = 256; g.Kpad = 256;
    g.outGroup = 4; g.outGroupStride = 2048; g.act = 1;
    launch_gemm(g);
  }
  // GEMM2: h_full[:, 1024:] = relu(edge_feat[edge_idxs] @ W_e)
  {
    GemmArgs g{};
    g.a0 = edge_feat; g.i0 = edge_idxs; g.c0 = 172; g.ld0 = 172;
    g.a1 = nullptr; g.i1 = nullptr; g.c1 = 0; g.ld1 = 0;
    g.tail = nullptr; g.taildiv = 1;
    g.bt = We_b; g.c = h_full + 1024;
    g.M = B; g.N = 1024; g.Kpad = 192;
    g.outGroup = 1; g.outGroupStride = 2048; g.act = 1;
    launch_gemm(g);
  }
  // GEMM3/4: up = tanh([emb[idx], h_full] @ W_uc)
  for (int side = 0; side < 2; ++side) {
    GemmArgs g{};
    g.a0 = emb; g.i0 = side ? dst_idxs : src_idxs; g.c0 = 256; g.ld0 = 256;
    g.a1 = h_full; g.i1 = nullptr; g.c1 = 2048; g.ld1 = 2048;
    g.tail = nullptr; g.taildiv = 1;
    g.bt = Wuc_b; g.c = side ? up_dst : up_src;
    g.M = B; g.N = 256; g.Kpad = 2304;
    g.outGroup = 1; g.outGroupStride = 256; g.act = 2;
    launch_gemm(g);
  }
  // GEMM5: h1 = h_full @ W_p
  {
    GemmArgs g{};
    g.a0 = h_full; g.i0 = nullptr; g.c0 = 2048; g.ld0 = 2048;
    g.a1 = nullptr; g.i1 = nullptr; g.c1 = 0; g.ld1 = 0;
    g.tail = nullptr; g.taildiv = 1;
    g.bt = Wp_b; g.c = h1;
    g.M = B; g.N = 256; g.Kpad = 2048;
    g.outGroup = 1; g.outGroupStride = 256; g.act = 0;
    launch_gemm(g);
  }

  // ---- scatter updated rows (src first, then dst: dst wins on collision) ----
  scatter_rows<<<dim3(B), dim3(256), 0, stream>>>(mem, src_idxs, up_src);
  scatter_rows<<<dim3(B), dim3(256), 0, stream>>>(mem, dst_idxs, up_dst);

  // ---- two sequential update phases ----
  for (int ph = 0; ph < 2; ++ph) {
    const int* flat   = ph ? unbrs_dst : unbrs_src;
    const float* uins = ph ? u_ins_dst : u_ins_src;
    const float* udel = ph ? u_del_dst : u_del_src;

    upd_att_kernel<<<dim3(B), dim3(256), 0, stream>>>(
        mem, h1, timestamps, flat, uins, udel, changed);

    // y = relu([changed, mem[flat], flag] @ W_1)
    {
      GemmArgs g{};
      g.a0 = changed; g.i0 = nullptr; g.c0 = 256; g.ld0 = 256;
      g.a1 = mem; g.i1 = flat; g.c1 = 256; g.ld1 = 256;
      g.tail = flags; g.taildiv = KU;
      g.bt = W1_b; g.c = ybuf;
      g.M = B * KU; g.N = 256; g.Kpad = 544;
      g.outGroup = 1; g.outGroupStride = 256; g.act = 1;
      launch_gemm(g);
    }
    // changed2 = relu([mem[flat], changed, flag] @ W_un)
    {
      GemmArgs g{};
      g.a0 = mem; g.i0 = flat; g.c0 = 256; g.ld0 = 256;
      g.a1 = changed; g.i1 = nullptr; g.c1 = 256; g.ld1 = 256;
      g.tail = flags; g.taildiv = KU;
      g.bt = Wun_b; g.c = chg2;
      g.M = B * KU; g.N = 256; g.Kpad = 544;
      g.outGroup = 1; g.outGroupStride = 256; g.act = 1;
      launch_gemm(g);
    }
    upd_apply_kernel<<<dim3(B * KU), dim3(256), 0, stream>>>(mem, ybuf, W_2, chg2, flat);
  }

  // ---- final link probabilities ----
  final_kernel<<<dim3(B), dim3(256), 0, stream>>>(
      mem, src_idxs, dst_idxs, neg_src_idxs, neg_dst_idxs, (float*)d_out, B);
}
